// ScaledDotProductAttention_197568495695
// MI455X (gfx1250) — matrix operand order repaired and checked by reading, not compiled, not run on hardware
//
#include <hip/hip_runtime.h>
#include <math.h>

typedef __attribute__((ext_vector_type(16))) _Float16 v16h;
typedef __attribute__((ext_vector_type(8)))  float    v8f;

#define S_LEN 2048
#define D_DIM 64
#define INV_T 0.125f   // 1/temperature, folded into q during f32->f16 conversion

// Issue this thread's slice of an 8KB contiguous global->LDS async copy.
// 256 threads * 2 * 16B = 8KB. Tracked by ASYNCcnt (CDNA5 async-to-LDS path).
__device__ __forceinline__ void async_stage_8k(const float* gbase, unsigned lds_base, int tid)
{
    #pragma unroll
    for (int j = 0; j < 2; ++j) {
        unsigned go = (unsigned)(tid + 256 * j) * 16u;   // byte offset, coalesced b128
        unsigned lo = lds_base + go;
        asm volatile("global_load_async_to_lds_b128 %0, %1, %2"
                     :: "v"(lo), "v"(go), "s"(gbase)
                     : "memory");
    }
}

__global__ __launch_bounds__(256)
void fa_wmma_kernel(const float* __restrict__ q,
                    const float* __restrict__ k,
                    const float* __restrict__ v,
                    const float* __restrict__ mask,
                    float* __restrict__ out)
{
    // f32 double buffers filled by async DMA; f16 tiles converted once per block
    __shared__ float    ldsK[2][32][D_DIM];
    __shared__ float    ldsV[2][32][D_DIM];
    __shared__ _Float16 ldsKh[32][72];     // [key][d], padded row (144B, 16B-aligned)
    __shared__ _Float16 ldsVt[D_DIM][40];  // [d][key], padded row (80B, 16B-aligned)
    __shared__ _Float16 ldsP[8][16][32];   // per-wave P tile

    const int tid  = threadIdx.x;
    const int lane = tid & 31;
    const int wv   = tid >> 5;        // wave id 0..7
    const int n    = lane & 15;       // column within 16-wide tile (B/C layout)
    const int hi   = lane >> 4;       // half-wave select

    const int bh = blockIdx.y;                     // 0..B*H-1
    const int q0 = blockIdx.x * 128 + wv * 16;     // this wave's 16 q rows

    const float* qp = q   + (size_t)bh * S_LEN * D_DIM;
    const float* kp = k   + (size_t)bh * S_LEN * D_DIM;
    const float* vp = v   + (size_t)bh * S_LEN * D_DIM;
    float*       op = out + (size_t)bh * S_LEN * D_DIM;

    const unsigned ldsK0 = (unsigned)(uintptr_t)&ldsK[0][0][0];
    const unsigned ldsK1 = (unsigned)(uintptr_t)&ldsK[1][0][0];
    const unsigned ldsV0 = (unsigned)(uintptr_t)&ldsV[0][0][0];
    const unsigned ldsV1 = (unsigned)(uintptr_t)&ldsV[1][0][0];

    // ---- Q as two A-fragments (16x32 f16 each): d 0..31 and d 32..63 ----
    // A layout: lane M = lane&15; slot s -> K = hi*16 + s*2 (+pair)
    v16h qa0, qa1;
    {
        const float* qrow = qp + (size_t)(q0 + n) * D_DIM;
        #pragma unroll
        for (int s = 0; s < 8; ++s) {
            int kk = (hi << 4) + (s << 1);
            float2 f0 = *(const float2*)(qrow + kk);
            float2 f1 = *(const float2*)(qrow + 32 + kk);
            qa0[2*s]   = (_Float16)(f0.x * INV_T);
            qa0[2*s+1] = (_Float16)(f0.y * INV_T);
            qa1[2*s]   = (_Float16)(f1.x * INV_T);
            qa1[2*s+1] = (_Float16)(f1.y * INV_T);
        }
    }

    // all-ones B fragment: row sums of P via one extra WMMA
    v16h onesf;
    #pragma unroll
    for (int i = 0; i < 16; ++i) onesf[i] = (_Float16)1.0f;

    // prefetch first K/V chunk into buffer 0
    async_stage_8k(kp, ldsK0, tid);
    async_stage_8k(vp, ldsV0, tid);

    v8f o0 = {}, o1 = {}, o2 = {}, o3 = {};   // 16 rows x 64 cols f32 accum
    float mrun[8], lrun[8];
    #pragma unroll
    for (int r = 0; r < 8; ++r) { mrun[r] = -__builtin_inff(); lrun[r] = 0.f; }

    for (int kb = 0; kb < S_LEN; kb += 32) {
        const int ib = (kb >> 5) & 1;

        // my DMA done; barrier => everyone's DMA done AND everyone finished
        // reading the buffers the next phases will overwrite (2-deep pipeline)
        asm volatile("s_wait_asynccnt 0x0" ::: "memory");
        __syncthreads();

        if (kb + 32 < S_LEN) {
            const float* knext = kp + (size_t)(kb + 32) * D_DIM;
            const float* vnext = vp + (size_t)(kb + 32) * D_DIM;
            async_stage_8k(knext, ib ? ldsK0 : ldsK1, tid);
            async_stage_8k(vnext, ib ? ldsV0 : ldsV1, tid);
        }

        // ---- cooperative f32 -> f16 convert pass (each element converted once) ----
        {   // K: keep [key][d]; 8 consecutive d per thread
            int row = tid >> 3;            // 0..31
            int c0  = (tid & 7) << 3;      // 0..56
            const float4* src = (const float4*)&ldsK[ib][row][c0];
            float4 x = src[0], y = src[1];
            union { _Float16 h[8]; uint4 u; } kk;
            kk.h[0] = (_Float16)x.x; kk.h[1] = (_Float16)x.y;
            kk.h[2] = (_Float16)x.z; kk.h[3] = (_Float16)x.w;
            kk.h[4] = (_Float16)y.x; kk.h[5] = (_Float16)y.y;
            kk.h[6] = (_Float16)y.z; kk.h[7] = (_Float16)y.w;
            *(uint4*)&ldsKh[row][c0] = kk.u;
        }
        {   // V: transpose to [d][key]; 8 consecutive keys per thread
            int d  = tid & 63;
            int k0 = (tid >> 6) << 3;      // 0,8,16,24
            union { _Float16 h[8]; uint4 u; } vvv;
            #pragma unroll
            for (int j = 0; j < 8; ++j)
                vvv.h[j] = (_Float16)ldsV[ib][k0 + j][d];
            *(uint4*)&ldsVt[d][k0] = vvv.u;
        }
        __syncthreads();

        // ---- K^T B-fragments: 16 consecutive f16 of Kh row -> v16h, no VALU ----
        v16h kf[2][2];
        #pragma unroll
        for (int t = 0; t < 2; ++t) {
            #pragma unroll
            for (int c = 0; c < 2; ++c) {
                union { v16h h; uint4 u[2]; } kb_;
                const uint4* p = (const uint4*)&ldsKh[t*16 + n][c*32 + (hi << 4)];
                kb_.u[0] = p[0]; kb_.u[1] = p[1];
                kf[t][c] = kb_.h;
            }
        }
        v8f c0 = {}, c1 = {};
        c0 = __builtin_amdgcn_wmma_f32_16x16x32_f16(false, qa0, false, kf[0][0], (short)0, c0, false, false);
        c0 = __builtin_amdgcn_wmma_f32_16x16x32_f16(false, qa1, false, kf[0][1], (short)0, c0, false, false);
        c1 = __builtin_amdgcn_wmma_f32_16x16x32_f16(false, qa0, false, kf[1][0], (short)0, c1, false, false);
        c1 = __builtin_amdgcn_wmma_f32_16x16x32_f16(false, qa1, false, kf[1][1], (short)0, c1, false, false);

        // ---- multiplicative mask + online softmax (row = r + 8*hi) ----
        float alpha_a[8];
        #pragma unroll
        for (int r = 0; r < 8; ++r) {
            const int row = q0 + r + (hi << 3);
            float m0 = mask[(size_t)row * S_LEN + kb + n];
            float m1 = mask[(size_t)row * S_LEN + kb + 16 + n];
            float s0 = c0[r] * m0;
            float s1 = c1[r] * m1;

            float mx = fmaxf(s0, s1);
            #pragma unroll
            for (int off = 1; off < 16; off <<= 1)
                mx = fmaxf(mx, __shfl_xor(mx, off, 32));
            float mn    = fmaxf(mrun[r], mx);
            alpha_a[r]  = __expf(mrun[r] - mn);
            mrun[r] = mn;

            float p0 = __expf(s0 - mn);
            float p1 = __expf(s1 - mn);

            // stash P into per-wave LDS tile (row-major 16x32 f16)
            ldsP[wv][r + (hi << 3)][n]      = (_Float16)p0;
            ldsP[wv][r + (hi << 3)][16 + n] = (_Float16)p1;

            o0[r] *= alpha_a[r]; o1[r] *= alpha_a[r];
            o2[r] *= alpha_a[r]; o3[r] *= alpha_a[r];
        }

        // ---- reload P as A-fragment: two contiguous 16B runs per lane ----
        v16h pf;
        {
            union { v16h h; uint4 u[2]; } pu;
            pu.u[0] = *(const uint4*)&ldsP[wv][n][hi << 4];        // K = hi*16 .. +7
            pu.u[1] = *(const uint4*)&ldsP[wv][n][(hi << 4) + 8];  // K = hi*16+8 .. +7
            pf = pu.h;
        }

        // ---- row sums of P via WMMA with all-ones B (replaces shuffle reduce) ----
        v8f ls = {};
        ls = __builtin_amdgcn_wmma_f32_16x16x32_f16(false, pf, false, onesf, (short)0, ls, false, false);
        #pragma unroll
        for (int r = 0; r < 8; ++r)
            lrun[r] = lrun[r] * alpha_a[r] + ls[r];

        // ---- V B-fragments: 16 consecutive f16 of Vt row -> v16h, no VALU ----
        v16h vf[4];
        #pragma unroll
        for (int t = 0; t < 4; ++t) {
            union { v16h h; uint4 u[2]; } vb_;
            const uint4* p = (const uint4*)&ldsVt[t*16 + n][hi << 4];
            vb_.u[0] = p[0]; vb_.u[1] = p[1];
            vf[t] = vb_.h;
        }
        o0 = __builtin_amdgcn_wmma_f32_16x16x32_f16(false, pf, false, vf[0], (short)0, o0, false, false);
        o1 = __builtin_amdgcn_wmma_f32_16x16x32_f16(false, pf, false, vf[1], (short)0, o1, false, false);
        o2 = __builtin_amdgcn_wmma_f32_16x16x32_f16(false, pf, false, vf[2], (short)0, o2, false, false);
        o3 = __builtin_amdgcn_wmma_f32_16x16x32_f16(false, pf, false, vf[3], (short)0, o3, false, false);
    }

    // ---- normalize and store (coalesced across each half-wave) ----
    #pragma unroll
    for (int r = 0; r < 8; ++r) {
        const int row = q0 + r + (hi << 3);
        float inv = 1.f / lrun[r];
        float* orow = op + (size_t)row * D_DIM + n;
        orow[0]  = o0[r] * inv;
        orow[16] = o1[r] * inv;
        orow[32] = o2[r] * inv;
        orow[48] = o3[r] * inv;
    }
}

extern "C" void kernel_launch(void* const* d_in, const int* in_sizes, int n_in,
                              void* d_out, int out_size, void* d_ws, size_t ws_size,
                              hipStream_t stream) {
    const float* q    = (const float*)d_in[0];
    const float* k    = (const float*)d_in[1];
    const float* v    = (const float*)d_in[2];
    const float* mask = (const float*)d_in[3];
    float* out = (float*)d_out;

    // grid: (S/128 q-tiles, B*H); block: 256 threads = 8 wave32 = 128 q rows
    dim3 grid(S_LEN / 128, 64);
    fa_wmma_kernel<<<grid, 256, 0, stream>>>(q, k, v, mask, out);
}
